// DiffusionNCA_fft2_hypernet_12575664243284
// MI455X (gfx1250) — compile-verified, matrix operand order checked by reading, not run on hardware
//
#include <hip/hip_runtime.h>
#include <hip/hip_bf16.h>
#include <math.h>

// ---------------- types for WMMA ----------------
typedef _Float16 v16h __attribute__((ext_vector_type(16)));
typedef _Float16 v8h  __attribute__((ext_vector_type(8)));
typedef float    v8f  __attribute__((ext_vector_type(8)));

union HFrag { v16h v; v8h h[2]; };

static __device__ __forceinline__ v8f wmma16(const HFrag& a, const HFrag& b, v8f c) {
  // D = A(16x32 f16) * B(32x16 f16) + C(16x16 f32)
  return __builtin_amdgcn_wmma_f32_16x16x32_f16(false, a.v, false, b.v, (short)0, c, false, false);
}

// ---------------- problem constants ----------------
#define BB   32
#define CC   32
#define EE   4
#define CH   36      // C+E
#define YC   72      // 2*(C+E)
#define KPAD 96      // YC padded to multiple of 32
#define HIDN 256
#define HH   128
#define WWW  128
#define HWP  16384   // H*W
#define INL  129     // hypernet latent (128 + cond)
#define HCS  132     // padded latent stride

#define R_CONV 1764            // (C+E)*49
#define R_F0   18432           // HID*72
#define R_F1   8192            // C*HID
#define R_TOT  (R_CONV + R_F0 + R_F1)

// ---------------- workspace layout (bytes, 256-aligned) ----------------
#define OFF_HCAT  0u
#define OFF_WCONV 16896u
#define OFF_W0    242688u
#define OFF_W1    1815552u
#define OFF_DX    2339840u
#define OFF_YH    77837312u

// =====================================================================
// Kernel 1: hypernet trunk -> hcat[B][132]  (h2[128], cond, 0,0,0)
// =====================================================================
__global__ __launch_bounds__(128) void k_trunk(
    const float* __restrict__ cond,
    const float* __restrict__ hv1, const float* __restrict__ hg1, const float* __restrict__ hb1,
    const float* __restrict__ hv2, const float* __restrict__ hg2, const float* __restrict__ hb2,
    float* __restrict__ hcat) {
  const int b = blockIdx.x;
  const int j = threadIdx.x;            // 0..127
  __shared__ float sh1[INL];
  const float c0 = cond[b];
  {
    float v = hv1[j];                   // IN == 1
    float inv = hg1[j] * rsqrtf(v * v + 1e-30f);
    float o = c0 * v * inv + hb1[j];
    sh1[j] = o > 0.f ? o : 0.f;
  }
  if (j == 0) sh1[128] = c0;
  __syncthreads();
  const float* v = hv2 + j * INL;
  float n2 = 0.f, dot = 0.f;
  for (int i = 0; i < INL; ++i) { float vv = v[i]; n2 += vv * vv; dot += vv * sh1[i]; }
  float o = dot * (hg2[j] * rsqrtf(n2 + 1e-30f)) + hb2[j];
  o = o > 0.f ? o : 0.f;
  hcat[b * HCS + j] = o;
  if (j == 0) {
    hcat[b * HCS + 128] = c0;
    hcat[b * HCS + 129] = 0.f; hcat[b * HCS + 130] = 0.f; hcat[b * HCS + 131] = 0.f;
  }
}

// =====================================================================
// Kernel 2: weight-norm heads. One thread = one output row (all batches).
//   conv -> f32  wconv[B][36][49]
//   fc0  -> f16  w0h[B][256][96]   (cols 72..95 zeroed)
//   fc1  -> f16  w1h[B][32][256]
// =====================================================================
__global__ __launch_bounds__(256) void k_heads(
    const float* __restrict__ hcat,
    const float* __restrict__ cv,  const float* __restrict__ cg,  const float* __restrict__ cb,
    const float* __restrict__ f0v, const float* __restrict__ f0g, const float* __restrict__ f0b,
    const float* __restrict__ f1v, const float* __restrict__ f1g, const float* __restrict__ f1b,
    float* __restrict__ wconv, _Float16* __restrict__ w0h, _Float16* __restrict__ w1h) {
  __shared__ float sh[BB * HCS];
  const int tid = threadIdx.x;
  for (int i = tid; i < BB * HCS; i += 256) sh[i] = hcat[i];
  __syncthreads();

  const int r = blockIdx.x * 256 + tid;
  if (r >= R_TOT) return;

  const float* v; float g, bias; int head, rr;
  if (r < R_CONV)            { head = 0; rr = r;                 v = cv  + rr * INL; g = cg[rr];  bias = cb[rr];  }
  else if (r < R_CONV + R_F0){ head = 1; rr = r - R_CONV;        v = f0v + rr * INL; g = f0g[rr]; bias = f0b[rr]; }
  else                       { head = 2; rr = r - R_CONV - R_F0; v = f1v + rr * INL; g = f1g[rr]; bias = f1b[rr]; }

  float n2 = 0.f;
  for (int i = 0; i < INL; ++i) { float vv = v[i]; n2 += vv * vv; }
  const float inv = g * rsqrtf(n2 + 1e-30f);

  for (int b = 0; b < BB; ++b) {
    const float* h = sh + b * HCS;
    float dot = 0.f;
    for (int i = 0; i < INL; ++i) dot += h[i] * v[i];
    const float val = dot * inv + bias;
    if (head == 0) {
      int ch = rr / 49, kk = rr % 49;
      wconv[((size_t)b * CH + ch) * 49 + kk] = val;
    } else if (head == 1) {
      int m = rr / YC, k = rr % YC;
      _Float16* p = w0h + ((size_t)b * HIDN + m) * KPAD;
      p[k] = (_Float16)val;
      if (k < KPAD - YC) p[YC + k] = (_Float16)0.f;   // zero pad cols 72..95
    } else {
      int m = rr / HIDN, k = rr % HIDN;
      w1h[((size_t)b * CC + m) * HIDN + k] = (_Float16)val;
    }
  }
}

// =====================================================================
// Kernel 3: positional encoding + 1x1 convs -> dx[B][36][HW] (f32)
// =====================================================================
__global__ __launch_bounds__(256) void k_ptdx(
    const float* __restrict__ x, const float* __restrict__ alive, const int* __restrict__ step,
    const float* __restrict__ pta_w, const float* __restrict__ pta_b,
    const float* __restrict__ ptb_w, const float* __restrict__ ptb_b,
    float* __restrict__ dx) {
  __shared__ float sA[32 * 16], sAb[32], sB[4 * 32], sBb[4];
  const int tid = threadIdx.x;
  for (int i = tid; i < 512; i += 256) sA[i] = pta_w[i];
  if (tid < 32)  sAb[tid] = pta_b[tid];
  if (tid < 128) sB[tid]  = ptb_w[tid];
  if (tid < 4)   sBb[tid] = ptb_b[tid];
  __syncthreads();

  const int gid = blockIdx.x * 256 + tid;
  const int b   = gid >> 14;
  const int rem = gid & (HWP - 1);
  const int hh  = rem >> 7;
  const int wc  = rem & (WWW - 1);

  float pte[4];
  pte[0] = 1.f - (float)wc * (1.f / (WWW - 1));
  pte[1] = 1.f - (float)hh * (1.f / (HH - 1));
  pte[2] = alive[b];
  pte[3] = (float)step[0];

  // emb[i] = [cos(i*1), cos(i*0.01), sin(i*1), sin(i*0.01)]
  float enc[16];
  #pragma unroll
  for (int i = 0; i < 4; ++i) {
    float fi = (float)i;
    enc[i * 4 + 0] = pte[i] * cosf(fi);
    enc[i * 4 + 1] = pte[i] * cosf(0.01f * fi);
    enc[i * 4 + 2] = pte[i] * sinf(fi);
    enc[i * 4 + 3] = pte[i] * sinf(0.01f * fi);
  }

  float pt[EE] = {0.f, 0.f, 0.f, 0.f};
  #pragma unroll 4
  for (int o = 0; o < 32; ++o) {
    float t = sAb[o];
    #pragma unroll
    for (int j = 0; j < 16; ++j) t += sA[o * 16 + j] * enc[j];
    float a = t / (1.f + expf(-t));       // SiLU
    #pragma unroll
    for (int e = 0; e < EE; ++e) pt[e] += sB[e * 32 + o] * a;
  }

  float* dxp = dx + (size_t)b * CH * HWP + rem;
  const float* xp = x + (size_t)b * CC * HWP + rem;
  #pragma unroll 8
  for (int c = 0; c < CC; ++c) dxp[(size_t)c * HWP] = xp[(size_t)c * HWP];
  #pragma unroll
  for (int e = 0; e < EE; ++e) dxp[(size_t)(CC + e) * HWP] = pt[e] + sBb[e];
}

// =====================================================================
// Kernel 4: per-sample depthwise 7x7 conv + PixelNorm -> yh[B][72][HW] f16
//   block: 8x32 spatial tile, loops 36 channels through an LDS halo tile.
// =====================================================================
#define TY 8
#define TX 32
#define TLW (TX + 6)
#define TLH (TY + 6)

__global__ __launch_bounds__(256) void k_conv(
    const float* __restrict__ dx, const float* __restrict__ wconv,
    _Float16* __restrict__ yh) {
  __shared__ float tile[TLH * TLW];
  __shared__ float wsh[49];
  __shared__ _Float16 stash[YC * 256];

  const int b  = blockIdx.z;
  const int h0 = blockIdx.y * TY;
  const int w0 = blockIdx.x * TX;
  const int tid = threadIdx.x;
  const int ty = tid >> 5, tx = tid & 31;

  float acc = 0.f;
  for (int ch = 0; ch < CH; ++ch) {
    const float* src = dx + ((size_t)b * CH + ch) * HWP;
    for (int i = tid; i < TLH * TLW; i += 256) {
      int r = i / TLW, c = i % TLW;
      int hh = h0 + r - 3, ww = w0 + c - 3;
      float v = 0.f;
      if (hh >= 0 && hh < HH && ww >= 0 && ww < WWW) v = src[hh * WWW + ww];
      tile[i] = v;
    }
    if (tid < 49) wsh[tid] = wconv[((size_t)b * CH + ch) * 49 + tid];
    __syncthreads();

    float s = 0.f;
    #pragma unroll
    for (int ky = 0; ky < 7; ++ky)
      #pragma unroll
      for (int kx = 0; kx < 7; ++kx)
        s += tile[(ty + ky) * TLW + (tx + kx)] * wsh[ky * 7 + kx];
    float dc = tile[(ty + 3) * TLW + (tx + 3)];
    acc += dc * dc + s * s;
    stash[ch * 256 + tid]        = (_Float16)dc;   // y channels 0..35 = dx
    stash[(CH + ch) * 256 + tid] = (_Float16)s;    // y channels 36..71 = conv
    __syncthreads();
  }

  const float inv = rsqrtf(acc * (1.f / YC) + 1e-8f);
  const int pix = (h0 + ty) * WWW + (w0 + tx);
  _Float16* dst = yh + (size_t)b * YC * HWP + pix;
  #pragma unroll 8
  for (int c = 0; c < YC; ++c)
    dst[(size_t)c * HWP] = (_Float16)((float)stash[c * 256 + tid] * inv);
}

// =====================================================================
// Kernel 5: WMMA double-GEMM + residual.
//   block = (batch b, 64-pixel tile). 8 waves.
//   GEMM1: hmid[256,64] = relu(W0[256,96] @ Y[96,64])   (K padded)
//   GEMM2: upd [32,64]  = W1[32,256] @ hmid
//   out = x + upd
// =====================================================================
__global__ __launch_bounds__(256) void k_gemm(
    const _Float16* __restrict__ yh, const _Float16* __restrict__ w0h,
    const _Float16* __restrict__ w1h, const float* __restrict__ x,
    float* __restrict__ out) {
  __shared__ _Float16 sY[64 * KPAD];     // [pixel][k]  12 KB
  __shared__ _Float16 sH[64 * HIDN];     // [pixel][k]  32 KB

  const int b   = blockIdx.y;
  const int p0  = blockIdx.x * 64;
  const int tid = threadIdx.x;
  const int wid = tid >> 5;
  const int lane = tid & 31;
  const int lh  = lane & 15;             // row/col within fragment
  const int half8 = (lane >> 4) << 3;    // 0 or 8: K sub-block per lane half

  // ---- stage Y tile (zero-padded K) ----
  for (int i = tid; i < KPAD * 64; i += 256) {
    int c = i >> 6, n = i & 63;
    _Float16 v = (_Float16)0.f;
    if (c < YC) v = yh[((size_t)b * YC + c) * HWP + p0 + n];
    sY[n * KPAD + c] = v;
  }
  __syncthreads();

  // ---- GEMM1: wave handles m-tiles {2w, 2w+1} x n-tiles 0..3 ----
  const int mt0 = wid * 2;
  v8f acc1[2][4];
  #pragma unroll
  for (int i = 0; i < 2; ++i)
    #pragma unroll
    for (int nt = 0; nt < 4; ++nt) acc1[i][nt] = {};

  #pragma unroll
  for (int ks = 0; ks < 3; ++ks) {
    const int k0 = ks * 32;
    HFrag A[2];
    #pragma unroll
    for (int i = 0; i < 2; ++i) {
      const _Float16* p = w0h + ((size_t)b * HIDN + (mt0 + i) * 16 + lh) * KPAD + k0 + half8;
      A[i].h[0] = *(const v8h*)p;
      A[i].h[1] = *(const v8h*)(p + 16);
    }
    #pragma unroll
    for (int nt = 0; nt < 4; ++nt) {
      HFrag Bf;
      const _Float16* q = &sY[(nt * 16 + lh) * KPAD + k0 + half8];
      Bf.h[0] = *(const v8h*)q;
      Bf.h[1] = *(const v8h*)(q + 16);
      acc1[0][nt] = wmma16(A[0], Bf, acc1[0][nt]);
      acc1[1][nt] = wmma16(A[1], Bf, acc1[1][nt]);
    }
  }

  // ---- relu + store hmid to LDS as f16 [n][k] ----
  #pragma unroll
  for (int i = 0; i < 2; ++i)
    #pragma unroll
    for (int nt = 0; nt < 4; ++nt)
      #pragma unroll
      for (int r = 0; r < 8; ++r) {
        float v = acc1[i][nt][r];
        v = v > 0.f ? v : 0.f;
        int m = (mt0 + i) * 16 + r + ((lane >> 4) << 3);
        int n = nt * 16 + lh;
        sH[n * HIDN + m] = (_Float16)v;
      }
  __syncthreads();

  // ---- GEMM2: wave w -> m-tile (w&1), n-tile (w>>1) ----
  const int mt = wid & 1;
  const int nt = wid >> 1;
  v8f acc2 = {};
  #pragma unroll
  for (int ks = 0; ks < 8; ++ks) {
    const int k0 = ks * 32;
    HFrag A, Bf;
    const _Float16* p = w1h + ((size_t)b * CC + mt * 16 + lh) * HIDN + k0 + half8;
    A.h[0] = *(const v8h*)p;
    A.h[1] = *(const v8h*)(p + 16);
    const _Float16* q = &sH[(nt * 16 + lh) * HIDN + k0 + half8];
    Bf.h[0] = *(const v8h*)q;
    Bf.h[1] = *(const v8h*)(q + 16);
    acc2 = wmma16(A, Bf, acc2);
  }

  // ---- residual epilogue ----
  #pragma unroll
  for (int r = 0; r < 8; ++r) {
    int c = mt * 16 + r + ((lane >> 4) << 3);
    int n = nt * 16 + lh;
    size_t idx = ((size_t)b * CC + c) * HWP + p0 + n;
    out[idx] = x[idx] + acc2[r];
  }
}

// =====================================================================
extern "C" void kernel_launch(void* const* d_in, const int* in_sizes, int n_in,
                              void* d_out, int out_size, void* d_ws, size_t ws_size,
                              hipStream_t stream) {
  const float* x     = (const float*)d_in[0];
  const float* cond  = (const float*)d_in[1];
  const float* alive = (const float*)d_in[2];
  const int*   step  = (const int*)  d_in[3];
  const float* hv1 = (const float*)d_in[4];
  const float* hg1 = (const float*)d_in[5];
  const float* hb1 = (const float*)d_in[6];
  const float* hv2 = (const float*)d_in[7];
  const float* hg2 = (const float*)d_in[8];
  const float* hb2 = (const float*)d_in[9];
  const float* cv  = (const float*)d_in[10];
  const float* cg  = (const float*)d_in[11];
  const float* cb  = (const float*)d_in[12];
  const float* f0v = (const float*)d_in[13];
  const float* f0g = (const float*)d_in[14];
  const float* f0b = (const float*)d_in[15];
  const float* f1v = (const float*)d_in[16];
  const float* f1g = (const float*)d_in[17];
  const float* f1b = (const float*)d_in[18];
  const float* pta_w = (const float*)d_in[19];
  const float* pta_b = (const float*)d_in[20];
  const float* ptb_w = (const float*)d_in[21];
  const float* ptb_b = (const float*)d_in[22];

  char* ws = (char*)d_ws;
  float*    hcat  = (float*)   (ws + OFF_HCAT);
  float*    wconv = (float*)   (ws + OFF_WCONV);
  _Float16* w0h   = (_Float16*)(ws + OFF_W0);
  _Float16* w1h   = (_Float16*)(ws + OFF_W1);
  float*    dx    = (float*)   (ws + OFF_DX);
  _Float16* yh    = (_Float16*)(ws + OFF_YH);
  float* out = (float*)d_out;

  k_trunk<<<BB, 128, 0, stream>>>(cond, hv1, hg1, hb1, hv2, hg2, hb2, hcat);
  k_heads<<<(R_TOT + 255) / 256, 256, 0, stream>>>(hcat, cv, cg, cb, f0v, f0g, f0b,
                                                   f1v, f1g, f1b, wconv, w0h, w1h);
  k_ptdx<<<(BB * HWP) / 256, 256, 0, stream>>>(x, alive, step, pta_w, pta_b, ptb_w, ptb_b, dx);
  k_conv<<<dim3(WWW / TX, HH / TY, BB), 256, 0, stream>>>(dx, wconv, yh);
  k_gemm<<<dim3(HWP / 64, BB), 256, 0, stream>>>(yh, w0h, w1h, x, out);
}
